// GraphTransformer_38611755991886
// MI455X (gfx1250) — matrix operand order repaired and checked by reading, not compiled, not run on hardware
//
#include <hip/hip_runtime.h>

// Graph-transformer edge-softmax attention for MI455X (gfx1250, wave32).
//   q:[N,H,C] k,v:[N,H,C] e:[E,H,C] f32, H=8, C=32, edge_dst sorted.
// Memory-bound (~1.3 GB -> ~56us @ 23.3 TB/s). One block per destination,
// one wave per head, lane = channel. QK^T dots via v_wmma_f32_16x16x32_f16
// on 16-edge tiles (flash-attention style online softmax per segment).
// Segment bounds + edge indices kept wave-uniform -> SALU/SMEM addressing.

#define HH 8
#define CC 32
#define EPSI 1e-6f
#define TILE 16
#define KPITCH 40   // halfs per ke row (80B, 16B aligned, bank-spread)

typedef _Float16 v16h __attribute__((ext_vector_type(16)));
typedef _Float16 v8h  __attribute__((ext_vector_type(8)));
typedef float    v8f  __attribute__((ext_vector_type(8)));

union AV { v16h v; struct { v8h lo, hi; } s; };

__device__ __forceinline__ float waveSum32(float v) {
#pragma unroll
    for (int off = 16; off >= 1; off >>= 1) v += __shfl_xor(v, off, 32);
    return v;
}

// ---- kernel 1: segment starts via binary search on sorted edge_dst ----
__global__ void seg_starts_kernel(const int* __restrict__ edge_dst, int E, int N,
                                  int* __restrict__ starts) {
    int n = blockIdx.x * blockDim.x + threadIdx.x;
    if (n > N) return;
    int lo = 0, hi = E;
    while (lo < hi) {
        int mid = (lo + hi) >> 1;
        if (edge_dst[mid] < n) lo = mid + 1; else hi = mid;
    }
    starts[n] = lo;
}

// ---- kernel 2: fused RMSNorm + gather + WMMA scores + online softmax ----
__global__ void __launch_bounds__(256)
graph_attn_kernel(const float* __restrict__ q, const float* __restrict__ k,
                  const float* __restrict__ v, const float* __restrict__ e,
                  const float* __restrict__ w_q, const float* __restrict__ w_k,
                  const int* __restrict__ edge_src,
                  const int* __restrict__ starts,
                  float* __restrict__ out) {
    __shared__ __attribute__((aligned(16))) _Float16 keT[HH][TILE][KPITCH]; // f16 KE tile
    __shared__ __attribute__((aligned(16))) float    veT[HH][TILE][CC];    // f32 VE tile
    __shared__ __attribute__((aligned(16))) _Float16 qnB[HH][CC];          // f16 qn

    const int n    = blockIdx.x;
    const int tid  = threadIdx.x;
    const int h    = tid >> 5;       // wave id == head
    const int lane = tid & 31;       // lane == channel
    const int half = lane >> 4;      // wave half for WMMA layouts
    const int mrow = lane & 15;

    // uniform addresses -> scalar (SMEM) loads; no shared mem, no barrier
    const int s0 = starts[n];
    const int s1 = starts[n + 1];

    const float wqv = w_q[lane];
    const float wkv = w_k[lane];
    const float qk_scale = 0.17677669529663687f; // 1/sqrt(32)

    // qn = rmsnorm(q)*w_q, fold in qk_scale; build WMMA B (all 16 cols = qn)
    {
        float qv  = q[(size_t)n * (HH * CC) + h * CC + lane];
        float qss = waveSum32(qv * qv);
        float qnv = qv * rsqrtf(qss * (1.0f / CC) + EPSI) * wqv * qk_scale;
        qnB[h][lane] = (_Float16)qnv;
    }
    AV b;
    b.s.lo = *(const v8h*)&qnB[h][half * 8];
    b.s.hi = *(const v8h*)&qnB[h][16 + half * 8];

    // zero VE tile once: padded rows must be finite (p=0 * NaN would poison acc)
#pragma unroll
    for (int j = 0; j < TILE; ++j) veT[h][j][lane] = 0.0f;

    float mRun = -INFINITY, lRun = 0.0f, accRun = 0.0f;

    for (int t0 = s0; t0 < s1; t0 += TILE) {
        const int T = (s1 - t0 < TILE) ? (s1 - t0) : TILE;

        // prefetch next tile's e stream (global_prefetch_b8; rare multi-tile path)
        if (t0 + TILE < s1)
            __builtin_prefetch(e + (size_t)(t0 + TILE) * (HH * CC) + h * CC + lane, 0, 1);

        // ---- stage tile: coalesced 128B row gathers, on-the-fly K rmsnorm ----
        for (int j = 0; j < T; ++j) {
            const int jj  = t0 + j;                                   // scalar
            const int src = __builtin_amdgcn_readfirstlane(edge_src[jj]);
            const float* __restrict__ krow = k + (size_t)src * (HH * CC) + h * CC;
            const float* __restrict__ vrow = v + (size_t)src * (HH * CC) + h * CC;
            const float* __restrict__ erow = e + (size_t)jj  * (HH * CC) + h * CC;
            const float kv = krow[lane];      // saddr + lane*4, 128B coalesced
            const float vv = vrow[lane];
            const float ev = erow[lane];
            const float kss = waveSum32(kv * kv);
            const float kev = kv * rsqrtf(kss * (1.0f / CC) + EPSI) * wkv + ev;
            keT[h][j][lane] = (_Float16)kev;   // f16 for WMMA A
            veT[h][j][lane] = vv + ev;         // f32 for PV accumulation
        }

        // ---- A in WMMA 16x32 f16 layout: two ds_load_b128 per lane ----
        AV a;
        const _Float16* rowp = &keT[h][mrow][0];
        a.s.lo = *(const v8h*)(rowp + half * 8);
        a.s.hi = *(const v8h*)(rowp + 16 + half * 8);

        v8f cz = {0.f, 0.f, 0.f, 0.f, 0.f, 0.f, 0.f, 0.f};
        v8f d = __builtin_amdgcn_wmma_f32_16x16x32_f16(
            false, a.v, false, b.v, (short)0, cz, false, false);

        // ---- D cols identical: lane holds s_{r + 8*half} in d[r] ----
        // mask padded rows to -inf (also kills stale-LDS NaN garbage)
        float sm[8];
#pragma unroll
        for (int r = 0; r < 8; ++r) {
            const int jIdx = r + (half << 3);
            sm[r] = (jIdx < T) ? d[r] : -INFINITY;
        }

        // tile max: 8 local max + one cross-half shfl
        float mt = sm[0];
#pragma unroll
        for (int r = 1; r < 8; ++r) mt = fmaxf(mt, sm[r]);
        mt = fmaxf(mt, __shfl_xor(mt, 16, 32));

        const float mNew = fmaxf(mRun, mt);
        const float corr = __expf(mRun - mNew);   // exp(-inf)=0 on first tile
        lRun *= corr; accRun *= corr;

        // p for own half; fetch other half's p with shfl_xor(16)
        float pw[8], po[8];
#pragma unroll
        for (int r = 0; r < 8; ++r) {
            pw[r] = __expf(sm[r] - mNew);         // padded -> exp(-inf)=0
            po[r] = __shfl_xor(pw[r], 16, 32);
        }
        // pl[j]=p_j (j<8), ph[j]=p_{j+8}, valid on every lane
        float lsum = 0.0f;
        float pl[8], ph[8];
#pragma unroll
        for (int r = 0; r < 8; ++r) {
            pl[r] = half ? po[r] : pw[r];
            ph[r] = half ? pw[r] : po[r];
            lsum += pl[r] + ph[r];
        }
        lRun += lsum;

        // fixed 16-step PV accumulation: broadcast row loads, p=0 on padding
#pragma unroll
        for (int j = 0; j < 8; ++j) accRun = fmaf(pl[j], veT[h][j][lane], accRun);
#pragma unroll
        for (int j = 0; j < 8; ++j) accRun = fmaf(ph[j], veT[h][j + 8][lane], accRun);

        mRun = mNew;
    }

    const float o = (lRun > 0.0f) ? accRun / fmaxf(lRun, 1e-30f) : 0.0f;
    out[(size_t)n * (HH * CC) + h * CC + lane] = o;
}

extern "C" void kernel_launch(void* const* d_in, const int* in_sizes, int n_in,
                              void* d_out, int out_size, void* d_ws, size_t ws_size,
                              hipStream_t stream) {
    const float* q   = (const float*)d_in[0];
    const float* k   = (const float*)d_in[1];
    const float* v   = (const float*)d_in[2];
    const float* e   = (const float*)d_in[3];
    const float* wq  = (const float*)d_in[4];
    const float* wk  = (const float*)d_in[5];
    const int* esrc  = (const int*)d_in[6];
    const int* edst  = (const int*)d_in[7];

    const int N = in_sizes[0] / (HH * CC);
    const int E = in_sizes[6];

    int* starts = (int*)d_ws;  // N+1 ints

    seg_starts_kernel<<<(N + 1 + 255) / 256, 256, 0, stream>>>(edst, E, N, starts);
    graph_attn_kernel<<<N, 256, 0, stream>>>(q, k, v, e, wq, wk, esrc, starts,
                                             (float*)d_out);
}
